// Attention_56281251447235
// MI455X (gfx1250) — matrix operand order repaired and checked by reading, not compiled, not run on hardware
//
#include <hip/hip_runtime.h>

// ---------------------------------------------------------------------------
// Multi-head attention forward for MI455X (gfx1250, wave32, WMMA).
// bf16 WMMA everywhere, fp32 accum.  Async global->LDS (ASYNCcnt) double-
// buffered staging for GEMM weight tiles and attention K/V tiles.
// ---------------------------------------------------------------------------

#define HEADS  16
#define DIMH   64
#define BATCH  4
#define SEQ    2048
#define DMODEL 1024
#define NQKV   3072
#define MTOT   (BATCH * SEQ)     // 8192
#define BH     (BATCH * HEADS)   // 64

typedef __attribute__((ext_vector_type(16))) __bf16 v16bf;
typedef __attribute__((ext_vector_type(8)))  float  v8f;

static __device__ __forceinline__ unsigned short f32_to_bf16(float f) {
  unsigned int u = __float_as_uint(f);
  u += 0x7FFFu + ((u >> 16) & 1u);          // round-to-nearest-even
  return (unsigned short)(u >> 16);
}

union FragU { uint4 q[2]; v16bf v; };

// A fragment: 16x32 bf16, row-major source, leading dim `ld` elements.
// ISA layout: lane row = lane&15; K elems: VGPR0-3 -> 8g..8g+7, VGPR4-7 -> 16+8g..
static __device__ __forceinline__ v16bf load_frag_A(const unsigned short* base,
                                                    int ld, int row, int k0, int g) {
  FragU u;
  const unsigned short* p = base + (size_t)row * ld + k0;
  u.q[0] = *(const uint4*)(p + 8 * g);
  u.q[1] = *(const uint4*)(p + 16 + 8 * g);
  return u.v;
}

// B fragment: 32x16 bf16 from a column-contiguous (transposed) source:
// lane col = lane&15; K elems: VGPR0-3 -> 8g..8g+7, VGPR4-7 -> 16+8g.. (same as A).
static __device__ __forceinline__ v16bf load_frag_B(const unsigned short* baseT,
                                                    int ld, int col, int k0, int g) {
  FragU u;
  const unsigned short* p = baseT + (size_t)col * ld + k0;
  u.q[0] = *(const uint4*)(p + 8 * g);
  u.q[1] = *(const uint4*)(p + 16 + 8 * g);
  return u.v;
}

#define WMMA_BF16(a, b, c) \
  __builtin_amdgcn_wmma_f32_16x16x32_bf16(false, (a), false, (b), (short)0, (c), false, false)

// gfx1250 async DMA: global -> LDS, 16B per lane, tracked by ASYNCcnt.
static __device__ __forceinline__ void async_copy_b128(unsigned lds_off, const void* gptr) {
  asm volatile("global_load_async_to_lds_b128 %0, %1, off"
               :: "v"(lds_off), "v"(gptr) : "memory");
}
static __device__ __forceinline__ void wait_async0() {
  asm volatile("s_wait_asynccnt 0x0" ::: "memory");
}
static __device__ __forceinline__ unsigned lds_off_of(const void* p) {
  return (unsigned)(size_t)p;   // low 32 bits of a generic LDS pointer = LDS byte offset
}

// ---------------------------------------------------------------------------
// K0a: fp32 -> bf16 elementwise
// ---------------------------------------------------------------------------
__global__ __launch_bounds__(256) void cvt_bf16_kernel(const float* __restrict__ src,
                                                       unsigned short* __restrict__ dst, int n) {
  int i = blockIdx.x * blockDim.x + threadIdx.x;
  if (i < n) dst[i] = f32_to_bf16(src[i]);
}

// K0b: transpose + convert: src[K][N] fp32 -> dstT[N][K] bf16 (K,N multiples of 16)
__global__ __launch_bounds__(256) void transpose_bf16_kernel(const float* __restrict__ src,
                                                             unsigned short* __restrict__ dstT,
                                                             int K, int N) {
  __shared__ float tile[16][17];
  int kb = blockIdx.y * 16, nb = blockIdx.x * 16;
  int tx = threadIdx.x, ty = threadIdx.y;
  tile[ty][tx] = src[(size_t)(kb + ty) * N + nb + tx];
  __syncthreads();
  dstT[(size_t)(nb + ty) * K + kb + tx] = f32_to_bf16(tile[tx][ty]);
}

// ---------------------------------------------------------------------------
// K1: QKV GEMM.  C[8192,3072] = xb @ Wqkv (wT[3072][1024] transposed weights).
// Block = 128 thr (4 waves), tile 128(M) x 64(N); each wave owns 2 row-subtiles.
// Weight tiles double-buffered in LDS via async global->LDS.
// Epilogue scatters q (scaled 1/8), k as [bh][n][64], v transposed [bh][64][n].
// ---------------------------------------------------------------------------
#define BT_STRIDE 40   // 80B rows (20 banks) -> conflict-free ds_load_b128

__global__ __launch_bounds__(128) void qkv_gemm_kernel(
    const unsigned short* __restrict__ xb, const unsigned short* __restrict__ wT,
    unsigned short* __restrict__ qb, unsigned short* __restrict__ kb,
    unsigned short* __restrict__ vtb) {
  __shared__ __attribute__((aligned(16))) unsigned short bts[2][64 * BT_STRIDE];
  const int tid = threadIdx.x;
  const int lane = tid & 31, wv = tid >> 5;
  const int g = lane >> 4, c = lane & 15;
  const int n0 = blockIdx.x * 64, m0 = blockIdx.y * 128;
  const int row0 = m0 + wv * 32 + c;        // subtile 0; subtile 1 = +16
  const int scol = tid >> 1, shalf = tid & 1;

  v8f acc[2][4] = {};

  // prologue: stage first weight tile
  async_copy_b128(lds_off_of(&bts[0][scol * BT_STRIDE + shalf * 16]),
                  wT + (size_t)(n0 + scol) * DMODEL + shalf * 16);
  int cur = 0;
  for (int k0 = 0; k0 < DMODEL; k0 += 32) {
    wait_async0();
    __syncthreads();                         // bts[cur] ready for everyone
    if (k0 + 32 < DMODEL)
      async_copy_b128(lds_off_of(&bts[cur ^ 1][scol * BT_STRIDE + shalf * 16]),
                      wT + (size_t)(n0 + scol) * DMODEL + (k0 + 32) + shalf * 16);
    v16bf a0 = load_frag_A(xb, DMODEL, row0, k0, g);
    v16bf a1 = load_frag_A(xb, DMODEL, row0 + 16, k0, g);
#pragma unroll
    for (int nt = 0; nt < 4; nt++) {
      v16bf b = load_frag_B(bts[cur], BT_STRIDE, nt * 16 + c, 0, g);
      acc[0][nt] = WMMA_BF16(a0, b, acc[0][nt]);
      acc[1][nt] = WMMA_BF16(a1, b, acc[1][nt]);
    }
    cur ^= 1;
  }

#pragma unroll
  for (int s = 0; s < 2; s++) {
#pragma unroll
    for (int nt = 0; nt < 4; nt++) {
      int col = n0 + nt * 16 + c;
      int sel = col >> 10;          // 0=q 1=k 2=v
      int cc = col & 1023;
      int h = cc >> 6, d = cc & 63;
#pragma unroll
      for (int r = 0; r < 8; r++) {
        int m = m0 + wv * 32 + s * 16 + r + 8 * g;
        int bb = m >> 11, nn = m & 2047;
        float v = acc[s][nt][r];
        if (sel == 0) {
          qb[((size_t)(bb * HEADS + h) * SEQ + nn) * DIMH + d] = f32_to_bf16(v * 0.125f);
        } else if (sel == 1) {
          kb[((size_t)(bb * HEADS + h) * SEQ + nn) * DIMH + d] = f32_to_bf16(v);
        } else {
          vtb[((size_t)(bb * HEADS + h) * DIMH + d) * SEQ + nn] = f32_to_bf16(v);
        }
      }
    }
  }
}

// ---------------------------------------------------------------------------
// K2: flash attention.  grid (SEQ/128, BH), block 256 (8 waves x 16 q-rows).
// K tile (32x64) and V^T tile (64x32) double-buffered in LDS via async DMA and
// shared by all 8 waves.  S = Q K^T (WMMA), online softmax in C/D layout,
// P restaged through padded LDS, O += P V via V^T (WMMA).
// ---------------------------------------------------------------------------
#define KT_STRIDE 72   // 144B rows (36 banks), 16B aligned
#define VT_STRIDE 40
#define P_STRIDE  40

__global__ __launch_bounds__(256) void attn_kernel(
    const unsigned short* __restrict__ qb, const unsigned short* __restrict__ kb,
    const unsigned short* __restrict__ vtb, unsigned short* __restrict__ attnb) {
  __shared__ __attribute__((aligned(16))) unsigned short kts[2][32 * KT_STRIDE];
  __shared__ __attribute__((aligned(16))) unsigned short vts[2][64 * VT_STRIDE];
  __shared__ __attribute__((aligned(16))) unsigned short pbuf[8 * 16 * P_STRIDE];
  const int tid = threadIdx.x;              // 0..255
  const int lane = tid & 31, wv = tid >> 5; // 8 waves
  const int g = lane >> 4, c = lane & 15;
  const int bh = blockIdx.y;
  const int i0 = blockIdx.x * 128 + wv * 16;

  const unsigned short* qbase = qb + (size_t)bh * SEQ * DIMH;
  const unsigned short* kbase = kb + (size_t)bh * SEQ * DIMH;
  const unsigned short* vtbase = vtb + (size_t)bh * DIMH * SEQ;
  unsigned short* pw = pbuf + wv * 16 * P_STRIDE;

  // staging decomposition: K tile = 32 rows x 8 b128-chunks, V tile = 64 x 4
  const int krow = tid >> 3, kch = tid & 7;
  const int vrow = tid >> 2, vch = tid & 3;

  // Q fragments (1/sqrt(d) folded in at QKV epilogue), d split 0..31 / 32..63
  v16bf aq0 = load_frag_A(qbase, DIMH, i0 + c, 0, g);
  v16bf aq1 = load_frag_A(qbase, DIMH, i0 + c, 32, g);

  v8f acc[4] = {};
  float m_i[8], l_i[8];
#pragma unroll
  for (int r = 0; r < 8; r++) { m_i[r] = -3.0e38f; l_i[r] = 0.0f; }

  // prologue: stage first K/V tile
  async_copy_b128(lds_off_of(&kts[0][krow * KT_STRIDE + kch * 8]),
                  kbase + (size_t)krow * DIMH + kch * 8);
  async_copy_b128(lds_off_of(&vts[0][vrow * VT_STRIDE + vch * 8]),
                  vtbase + (size_t)vrow * SEQ + vch * 8);
  int cur = 0;

  for (int j0 = 0; j0 < SEQ; j0 += 32) {
    wait_async0();
    __syncthreads();                        // tile[cur] fully staged, prev reads done
    if (j0 + 32 < SEQ) {
      async_copy_b128(lds_off_of(&kts[cur ^ 1][krow * KT_STRIDE + kch * 8]),
                      kbase + (size_t)(j0 + 32 + krow) * DIMH + kch * 8);
      async_copy_b128(lds_off_of(&vts[cur ^ 1][vrow * VT_STRIDE + vch * 8]),
                      vtbase + (size_t)vrow * SEQ + (j0 + 32) + vch * 8);
    }
    const unsigned short* kt = kts[cur];
    const unsigned short* vt = vts[cur];

    // ---- S = Q K^T for this 16x32 tile (two 16-col halves) ----
    v8f s0 = {}, s1 = {};
    {
      v16bf b00 = load_frag_B(kt, KT_STRIDE, c, 0, g);
      v16bf b01 = load_frag_B(kt, KT_STRIDE, c, 32, g);
      s0 = WMMA_BF16(aq0, b00, s0);
      s0 = WMMA_BF16(aq1, b01, s0);
      v16bf b10 = load_frag_B(kt, KT_STRIDE, 16 + c, 0, g);
      v16bf b11 = load_frag_B(kt, KT_STRIDE, 16 + c, 32, g);
      s1 = WMMA_BF16(aq0, b10, s1);
      s1 = WMMA_BF16(aq1, b11, s1);
    }

    // ---- online softmax: rows r+8g, cols lane&15 within 16-lane groups ----
#pragma unroll
    for (int r = 0; r < 8; r++) {
      float sv0 = s0[r], sv1 = s1[r];
      float mx = fmaxf(sv0, sv1);
      mx = fmaxf(mx, __shfl_xor(mx, 1, 16));
      mx = fmaxf(mx, __shfl_xor(mx, 2, 16));
      mx = fmaxf(mx, __shfl_xor(mx, 4, 16));
      mx = fmaxf(mx, __shfl_xor(mx, 8, 16));
      float mnew = fmaxf(m_i[r], mx);
      float alpha = __expf(m_i[r] - mnew);
      float p0 = __expf(sv0 - mnew);
      float p1 = __expf(sv1 - mnew);
      float ps = p0 + p1;
      ps += __shfl_xor(ps, 1, 16);
      ps += __shfl_xor(ps, 2, 16);
      ps += __shfl_xor(ps, 4, 16);
      ps += __shfl_xor(ps, 8, 16);
      l_i[r] = l_i[r] * alpha + ps;
      m_i[r] = mnew;
#pragma unroll
      for (int dt = 0; dt < 4; dt++) acc[dt][r] *= alpha;
      pw[(r + 8 * g) * P_STRIDE + c] = f32_to_bf16(p0);
      pw[(r + 8 * g) * P_STRIDE + 16 + c] = f32_to_bf16(p1);
    }
    __syncthreads();                        // P store -> P reload (cross-lane)

    // ---- reload P as A fragment, O += P V via V^T ----
    FragU up;
    up.q[0] = *(const uint4*)(pw + c * P_STRIDE + 8 * g);
    up.q[1] = *(const uint4*)(pw + c * P_STRIDE + 16 + 8 * g);
    v16bf ap = up.v;
#pragma unroll
    for (int dt = 0; dt < 4; dt++) {
      v16bf bv = load_frag_B(vt, VT_STRIDE, dt * 16 + c, 0, g);
      acc[dt] = WMMA_BF16(ap, bv, acc[dt]);
    }
    cur ^= 1;
  }

  // ---- epilogue: O /= rowsum, write bf16 to attn[b][n][h*64+d] ----
  const int bb = bh >> 4, h = bh & 15;
#pragma unroll
  for (int r = 0; r < 8; r++) {
    float rinv = 1.0f / l_i[r];
    int rowg = i0 + r + 8 * g;
    size_t base = ((size_t)bb * SEQ + rowg) * DMODEL + h * DIMH;
#pragma unroll
    for (int dt = 0; dt < 4; dt++)
      attnb[base + dt * 16 + c] = f32_to_bf16(acc[dt][r] * rinv);
  }
}

// ---------------------------------------------------------------------------
// K3: output projection + bias.  out[8192,1024] fp32 = attn_bf16 @ Wout + b.
// Same 128x64 async-staged WMMA GEMM as K1.
// ---------------------------------------------------------------------------
__global__ __launch_bounds__(128) void out_gemm_kernel(
    const unsigned short* __restrict__ ab, const unsigned short* __restrict__ wT,
    const float* __restrict__ bias, float* __restrict__ out) {
  __shared__ __attribute__((aligned(16))) unsigned short bts[2][64 * BT_STRIDE];
  const int tid = threadIdx.x;
  const int lane = tid & 31, wv = tid >> 5;
  const int g = lane >> 4, c = lane & 15;
  const int n0 = blockIdx.x * 64, m0 = blockIdx.y * 128;
  const int row0 = m0 + wv * 32 + c;
  const int scol = tid >> 1, shalf = tid & 1;

  v8f acc[2][4] = {};
  async_copy_b128(lds_off_of(&bts[0][scol * BT_STRIDE + shalf * 16]),
                  wT + (size_t)(n0 + scol) * DMODEL + shalf * 16);
  int cur = 0;
  for (int k0 = 0; k0 < DMODEL; k0 += 32) {
    wait_async0();
    __syncthreads();
    if (k0 + 32 < DMODEL)
      async_copy_b128(lds_off_of(&bts[cur ^ 1][scol * BT_STRIDE + shalf * 16]),
                      wT + (size_t)(n0 + scol) * DMODEL + (k0 + 32) + shalf * 16);
    v16bf a0 = load_frag_A(ab, DMODEL, row0, k0, g);
    v16bf a1 = load_frag_A(ab, DMODEL, row0 + 16, k0, g);
#pragma unroll
    for (int nt = 0; nt < 4; nt++) {
      v16bf b = load_frag_B(bts[cur], BT_STRIDE, nt * 16 + c, 0, g);
      acc[0][nt] = WMMA_BF16(a0, b, acc[0][nt]);
      acc[1][nt] = WMMA_BF16(a1, b, acc[1][nt]);
    }
    cur ^= 1;
  }

#pragma unroll
  for (int s = 0; s < 2; s++) {
#pragma unroll
    for (int nt = 0; nt < 4; nt++) {
      int col = n0 + nt * 16 + c;
      float bv = bias[col];
#pragma unroll
      for (int r = 0; r < 8; r++) {
        int m = m0 + wv * 32 + s * 16 + r + 8 * g;
        out[(size_t)m * DMODEL + col] = acc[s][nt][r] + bv;
      }
    }
  }
}

// ---------------------------------------------------------------------------
extern "C" void kernel_launch(void* const* d_in, const int* in_sizes, int n_in,
                              void* d_out, int out_size, void* d_ws, size_t ws_size,
                              hipStream_t stream) {
  (void)in_sizes; (void)n_in; (void)out_size; (void)ws_size;
  const float* x     = (const float*)d_in[0];   // [4,2048,1024]
  const float* w_qkv = (const float*)d_in[1];   // [1024,3072]
  const float* w_out = (const float*)d_in[2];   // [1024,1024]
  const float* b_out = (const float*)d_in[3];   // [1024]
  float* out = (float*)d_out;

  char* ws = (char*)d_ws;
  unsigned short* xb    = (unsigned short*)ws; ws += (size_t)MTOT * DMODEL * 2;
  unsigned short* wqkvT = (unsigned short*)ws; ws += (size_t)NQKV * DMODEL * 2;
  unsigned short* woutT = (unsigned short*)ws; ws += (size_t)DMODEL * DMODEL * 2;
  unsigned short* qbuf  = (unsigned short*)ws; ws += (size_t)BH * SEQ * DIMH * 2;
  unsigned short* kbuf  = (unsigned short*)ws; ws += (size_t)BH * SEQ * DIMH * 2;
  unsigned short* vtbuf = (unsigned short*)ws; ws += (size_t)BH * DIMH * SEQ * 2;
  unsigned short* attnb = (unsigned short*)ws; ws += (size_t)MTOT * DMODEL * 2;

  const int nx = MTOT * DMODEL;
  cvt_bf16_kernel<<<(nx + 255) / 256, 256, 0, stream>>>(x, xb, nx);
  transpose_bf16_kernel<<<dim3(NQKV / 16, DMODEL / 16), dim3(16, 16), 0, stream>>>(
      w_qkv, wqkvT, DMODEL, NQKV);
  transpose_bf16_kernel<<<dim3(DMODEL / 16, DMODEL / 16), dim3(16, 16), 0, stream>>>(
      w_out, woutT, DMODEL, DMODEL);

  qkv_gemm_kernel<<<dim3(NQKV / 64, MTOT / 128), 128, 0, stream>>>(xb, wqkvT, qbuf, kbuf, vtbuf);
  attn_kernel<<<dim3(SEQ / 128, BH), 256, 0, stream>>>(qbuf, kbuf, vtbuf, attnb);
  out_gemm_kernel<<<dim3(DMODEL / 64, MTOT / 128), 128, 0, stream>>>(attnb, woutT, b_out, out);
}
